// AutoformerEncoder_8538394984517
// MI455X (gfx1250) — compile-verified, matrix operand order checked by reading, not run on hardware
//
#include <hip/hip_runtime.h>
#include <hip/hip_bf16.h>
#include <math.h>

// ---------------- problem constants ----------------
#define BB 32
#define LL 4096
#define DD 512
#define KSZ 25
#define TOPK 16
#define LN_EPS 1e-5f
#define PI_F 3.14159265358979323846f

typedef __attribute__((ext_vector_type(16))) __bf16         v16bf;
typedef __attribute__((ext_vector_type(8)))  float          v8f;
typedef __attribute__((ext_vector_type(8)))  unsigned short u16x8;
typedef __attribute__((ext_vector_type(4)))  unsigned int   u32x4;

__device__ __forceinline__ unsigned short f2bf(float f) {
    unsigned int u = __float_as_uint(f);
    unsigned int r = u + 0x7FFFu + ((u >> 16) & 1u);   // round-to-nearest-even
    return (unsigned short)(r >> 16);
}

// ============================================================
// 1) Series decomposition: trend = movavg25(reflect-pad), seasonal = x - trend
// ============================================================
__global__ __launch_bounds__(256) void decomp_kernel(
    const float* __restrict__ x, float* __restrict__ trend,
    float* __restrict__ seasonal) {
    int i = blockIdx.x * blockDim.x + threadIdx.x;       // flat over B*L*D
    int c = i & (DD - 1);
    int l = (i >> 9) & (LL - 1);
    int b = i >> 21;
    const float* xb = x + (size_t)b * LL * DD + c;
    float s = 0.f;
#pragma unroll
    for (int t = 0; t < KSZ; ++t) {
        int lp = l + t - (KSZ / 2);
        lp = lp < 0 ? -lp : lp;                          // reflect (no edge repeat)
        lp = lp >= LL ? 2 * LL - 2 - lp : lp;
        s += xb[(size_t)lp * DD];
    }
    float tr = s * (1.0f / KSZ);
    trend[i] = tr;
    seasonal[i] = x[i] - tr;
}

// ============================================================
// 2) zero the sparse autocorr tensor (bf16 [B,L,D]) with b128 stores
// ============================================================
__global__ __launch_bounds__(256) void zero_kernel(u32x4* __restrict__ p, int n) {
    int i = blockIdx.x * blockDim.x + threadIdx.x;
    int stride = gridDim.x * blockDim.x;
    u32x4 z = 0;
    for (; i < n; i += stride) p[i] = z;
}

// ============================================================
// 3) per-(b,c) autocorrelation via in-LDS radix-2 FFT + top-16 lags
//    corr = (1/N) * FFT(|FFT(x)|^2).re   (power spectrum is real-even)
// ============================================================
__device__ __forceinline__ void fft4096_inplace(float* re, float* im, int tid) {
    // data must already be bit-reversal permuted; DIT, twiddle e^{-2pi i}
    for (int st = 1; st <= 12; ++st) {
        int len = 1 << st;
        int half = len >> 1;
        for (int u = tid; u < (LL >> 1); u += 256) {
            int pos = u & (half - 1);
            int base = (u >> (st - 1)) << st;
            float ang = -2.0f * PI_F * (float)pos / (float)len;
            float sn, cs;
            __sincosf(ang, &sn, &cs);
            int i0 = base + pos, i1 = i0 + half;
            float ar = re[i0], ai = im[i0];
            float br = re[i1], bi = im[i1];
            float tr = br * cs - bi * sn;
            float ti = br * sn + bi * cs;
            re[i0] = ar + tr; im[i0] = ai + ti;
            re[i1] = ar - tr; im[i1] = ai - ti;
        }
        __syncthreads();
    }
}

__global__ __launch_bounds__(256) void autocorr_topk_kernel(
    const float* __restrict__ seasonal, unsigned short* __restrict__ a /*bf16*/) {
    __shared__ float re[LL];
    __shared__ float im[LL];
    __shared__ float redV[256];
    __shared__ int   redI[256];
    __shared__ int   topI[TOPK];
    __shared__ float topV[TOPK];

    int bc = blockIdx.x;
    int c = bc & (DD - 1);
    int b = bc >> 9;
    int tid = threadIdx.x;
    const float* src = seasonal + (size_t)b * LL * DD + c;

    // load with bit-reversal (12-bit)
    for (int i = tid; i < LL; i += 256) {
        int j = (int)(__brev((unsigned)i) >> 20);
        re[j] = src[(size_t)i * DD];
        im[j] = 0.f;
    }
    __syncthreads();
    fft4096_inplace(re, im, tid);                 // X = FFT(x)

    // power spectrum (natural order), then bit-reverse permute in place
    for (int i = tid; i < LL; i += 256) {
        float r = re[i], m = im[i];
        re[i] = r * r + m * m;
        im[i] = 0.f;
    }
    __syncthreads();
    for (int i = tid; i < LL; i += 256) {
        int j = (int)(__brev((unsigned)i) >> 20);
        if (j > i) { float t = re[i]; re[i] = re[j]; re[j] = t; }
    }
    __syncthreads();
    fft4096_inplace(re, im, tid);                 // FFT(P) -> corr * N

    // corr + |corr| (|.| kept in im[] as selection key)
    for (int i = tid; i < LL; i += 256) {
        float v = re[i] * (1.0f / LL);
        re[i] = v;
        im[i] = fabsf(v);
    }
    __syncthreads();

    // iterative top-16 argmax
    for (int k = 0; k < TOPK; ++k) {
        float bv = -1.f; int bi = 0;
        for (int i = tid; i < LL; i += 256) {
            float v = im[i];
            if (v > bv) { bv = v; bi = i; }
        }
        redV[tid] = bv; redI[tid] = bi;
        __syncthreads();
        for (int s = 128; s > 0; s >>= 1) {
            if (tid < s && redV[tid + s] > redV[tid]) {
                redV[tid] = redV[tid + s]; redI[tid] = redI[tid + s];
            }
            __syncthreads();
        }
        if (tid == 0) {
            int idx = redI[0];
            topI[k] = idx;
            topV[k] = re[idx];
            im[idx] = -1.f;                        // exclude from next rounds
        }
        __syncthreads();
    }

    if (tid < TOPK) {
        int idx = topI[tid];
        a[((size_t)b * LL + idx) * DD + c] = f2bf(topV[tid]);
    }
}

// ============================================================
// 4) pre-swizzle W (fp32 [512,512]) into per-lane bf16 WMMA B fragments.
//    Fragment element e of lane ln in tile (kt,nt):
//      K = kt*32 + (ln>>4)*8 + (e<8 ? e : e+8),  N = nt*16 + (ln&15)
// ============================================================
__global__ __launch_bounds__(256) void swizzle_w_kernel(
    const float* __restrict__ w, unsigned short* __restrict__ wsw) {
    int g = blockIdx.x * blockDim.x + threadIdx.x;     // 16*32*32 = 16384 groups
    if (g >= 16 * 32 * 32) return;
    int lane = g & 31;
    int nt = (g >> 5) & 31;
    int kt = g >> 10;
    int khalf = (lane >> 4) << 3;
    int n = nt * 16 + (lane & 15);
    unsigned short out[16];
#pragma unroll
    for (int e = 0; e < 16; ++e) {
        int k = kt * 32 + khalf + (e < 8 ? e : e + 8);
        out[e] = f2bf(w[k * DD + n]);
    }
    u16x8* dst = (u16x8*)&wsw[(size_t)g * 16];
    dst[0] = *(u16x8*)&out[0];
    dst[1] = *(u16x8*)&out[8];
}

// ============================================================
// 5) fused bf16-WMMA GEMM (ff = a @ W + bias) + residual + LayerNorm
//    block = 256 threads = 8 wave32s; 16 rows x 512 cols per block.
// ============================================================
__global__ __launch_bounds__(256) void gemm_ln_kernel(
    const unsigned short* __restrict__ a,    // bf16 [B*L, 512]
    const unsigned short* __restrict__ wsw,  // swizzled bf16 W
    const float* __restrict__ bias,
    const float* __restrict__ gamma,
    const float* __restrict__ beta,
    float* __restrict__ seasonal)            // in/out [B*L, 512]
{
    __shared__ __attribute__((aligned(16))) unsigned short lds_a[16 * DD];
    __shared__ float lds_o[16 * DD];

    int tid = threadIdx.x;
    size_t row0 = (size_t)blockIdx.x * 16;

    // stage A chunk (16 rows x 512 bf16 = 16 KB) into LDS
    {
        const u16x8* ga = (const u16x8*)(a + row0 * DD);
        u16x8* la = (u16x8*)lds_a;
        for (int i = tid; i < 16 * DD / 8; i += 256) la[i] = ga[i];
    }
    __syncthreads();

    int wave = tid >> 5;            // 0..7
    int lane = tid & 31;
    int m = lane & 15;              // A-matrix row within tile
    int khalf = (lane >> 4) << 3;   // K sub-block select per ISA layout

    for (int nt4 = 0; nt4 < 4; ++nt4) {
        int nt = wave * 4 + nt4;    // column tile 0..31
        v8f acc = 0.f;
#pragma unroll 4
        for (int kt = 0; kt < 16; ++kt) {
            int koff = kt * 32 + khalf;
            u16x8 alo = *(const u16x8*)&lds_a[m * DD + koff];
            u16x8 ahi = *(const u16x8*)&lds_a[m * DD + koff + 16];
            const u16x8* bp =
                (const u16x8*)&wsw[((size_t)(kt * 32 + nt) * 32 + lane) * 16];
            if (kt + 1 < 16)  // prefetch next K-step B fragment (global_prefetch_b8)
                __builtin_prefetch(
                    &wsw[((size_t)((kt + 1) * 32 + nt) * 32 + lane) * 16], 0, 1);
            u16x8 blo = bp[0], bhi = bp[1];
            union { v16bf v; u16x8 h[2]; } af, bfr;
            af.h[0] = alo; af.h[1] = ahi;
            bfr.h[0] = blo; bfr.h[1] = bhi;
            acc = __builtin_amdgcn_wmma_f32_16x16x32_bf16(
                false, af.v, false, bfr.v, (short)0, acc, false, false);
        }
        // C/D layout: VGPR r -> M = r + 8*(lane>>4), N = lane&15
        int Mbase = (lane >> 4) * 8;
        int n = lane & 15;
#pragma unroll
        for (int r = 0; r < 8; ++r)
            lds_o[(Mbase + r) * DD + nt * 16 + n] = acc[r];
    }
    __syncthreads();

    // LayerNorm over D for each of the 16 rows: 16 threads per row,
    // cross-lane reduce with wave32 shfl_xor (width 16).
    int rrow = tid >> 4;
    int tr = tid & 15;
    float* srow = seasonal + (row0 + rrow) * DD;
    float vals[32];
    float sum = 0.f, ssq = 0.f;
#pragma unroll
    for (int j = 0; j < 32; ++j) {
        int d = tr + j * 16;
        float v = srow[d] + lds_o[rrow * DD + d] + bias[d];
        vals[j] = v;
        sum += v;
        ssq += v * v;
    }
    for (int off = 8; off > 0; off >>= 1) {
        sum += __shfl_xor(sum, off, 16);
        ssq += __shfl_xor(ssq, off, 16);
    }
    float mean = sum * (1.0f / DD);
    float var = ssq * (1.0f / DD) - mean * mean;
    float rs = rsqrtf(var + LN_EPS);
#pragma unroll
    for (int j = 0; j < 32; ++j) {
        int d = tr + j * 16;
        srow[d] = (vals[j] - mean) * rs * gamma[d] + beta[d];
    }
}

// ============================================================
// 6) final LN: out = LN(seasonal + trend)
// ============================================================
__global__ __launch_bounds__(256) void final_ln_kernel(
    const float* __restrict__ seasonal, const float* __restrict__ trend,
    const float* __restrict__ g, const float* __restrict__ bt,
    float* __restrict__ out) {
    int tid = threadIdx.x;
    size_t row = (size_t)blockIdx.x * 16 + (tid >> 4);
    int tr = tid & 15;
    const float* s = seasonal + row * DD;
    const float* t = trend + row * DD;
    float vals[32];
    float sum = 0.f, ssq = 0.f;
#pragma unroll
    for (int j = 0; j < 32; ++j) {
        int d = tr + j * 16;
        float v = s[d] + t[d];
        vals[j] = v;
        sum += v;
        ssq += v * v;
    }
    for (int off = 8; off > 0; off >>= 1) {
        sum += __shfl_xor(sum, off, 16);
        ssq += __shfl_xor(ssq, off, 16);
    }
    float mean = sum * (1.0f / DD);
    float var = ssq * (1.0f / DD) - mean * mean;
    float rs = rsqrtf(var + LN_EPS);
    float* o = out + row * DD;
#pragma unroll
    for (int j = 0; j < 32; ++j) {
        int d = tr + j * 16;
        o[d] = (vals[j] - mean) * rs * g[d] + bt[d];
    }
}

// ============================================================
// launch
// ============================================================
extern "C" void kernel_launch(void* const* d_in, const int* in_sizes, int n_in,
                              void* d_out, int out_size, void* d_ws, size_t ws_size,
                              hipStream_t stream) {
    (void)in_sizes; (void)n_in; (void)out_size; (void)ws_size;

    const float* x   = (const float*)d_in[0];
    const float* w0  = (const float*)d_in[1];
    const float* b0  = (const float*)d_in[2];
    const float* g0  = (const float*)d_in[3];
    const float* bt0 = (const float*)d_in[4];
    const float* w1  = (const float*)d_in[5];
    const float* b1  = (const float*)d_in[6];
    const float* g1  = (const float*)d_in[7];
    const float* bt1 = (const float*)d_in[8];
    const float* gf  = (const float*)d_in[9];
    const float* btf = (const float*)d_in[10];

    const size_t ND = (size_t)BB * LL * DD;              // 67,108,864
    float* trend    = (float*)d_ws;
    float* seasonal = trend + ND;
    unsigned short* abuf = (unsigned short*)(seasonal + ND);
    unsigned short* wsw0 = abuf + ND;
    unsigned short* wsw1 = wsw0 + 16 * 32 * 32 * 16;

    swizzle_w_kernel<<<64, 256, 0, stream>>>(w0, wsw0);
    swizzle_w_kernel<<<64, 256, 0, stream>>>(w1, wsw1);

    decomp_kernel<<<(unsigned)(ND / 256), 256, 0, stream>>>(x, trend, seasonal);

    const int nZero = (int)(ND / 8);                     // u32x4 = 8 bf16
    const float* biases[2] = {b0, b1};
    const float* gammas[2] = {g0, g1};
    const float* betas[2]  = {bt0, bt1};
    const unsigned short* wsws[2] = {wsw0, wsw1};

    for (int layer = 0; layer < 2; ++layer) {
        zero_kernel<<<8192, 256, 0, stream>>>((u32x4*)abuf, nZero);
        autocorr_topk_kernel<<<BB * DD, 256, 0, stream>>>(seasonal, abuf);
        gemm_ln_kernel<<<(BB * LL) / 16, 256, 0, stream>>>(
            abuf, wsws[layer], biases[layer], gammas[layer], betas[layer],
            seasonal);
    }

    final_ln_kernel<<<(BB * LL) / 16, 256, 0, stream>>>(
        seasonal, trend, gf, btf, (float*)d_out);
}
